// GATWithJK_16552803958873
// MI455X (gfx1250) — matrix operand order repaired and checked by reading, not compiled, not run on hardware
//
#include <hip/hip_runtime.h>
#include <hip/hip_bf16.h>

typedef __bf16 bf16_t;
typedef bf16_t v16bf __attribute__((ext_vector_type(16)));
typedef bf16_t v8bf  __attribute__((ext_vector_type(8)));
typedef float  v8f   __attribute__((ext_vector_type(8)));

// A 16x32 bf16 fragment (contiguous K, matches B):
//   lane half 0 -> K {0..15}, lane half 1 -> K {16..31}
__device__ __forceinline__ v16bf load_afrag(const bf16_t* __restrict__ arow, int hh) {
    v8bf lo = *(const v8bf*)(arow + hh * 16);
    v8bf hi = *(const v8bf*)(arow + hh * 16 + 8);
    v16bf a;
#pragma unroll
    for (int i = 0; i < 8; ++i) { a[i] = lo[i]; a[i + 8] = hi[i]; }
    return a;
}

// ---------------------------------------------------------------------------
// Split-K dual-source bf16 WMMA GEMM:
//   C[M,O] = concat_K(A1[M,K1], A2[M,K2]) @ B^T   (B row-major [O, K1+K2])
// Block = 64 threads (2 waves); each wave computes TWO adjacent 16x16 O-tiles
// sharing one A fragment (load:wmma = 3:2). Two sequential K phases keep the
// inner loops free of per-iteration source selects. Fused epilogue: +bias,
// ReLU, optional bf16 mirror at ldcbf. B fragment is 16 contiguous bf16 per
// lane (column-major B). K1,K2 mult of 32; M mult of 16; O mult of 64.
// EXEC all-ones around every WMMA.
// ---------------------------------------------------------------------------
__global__ void k_gemm_bf16(const bf16_t* __restrict__ A1, int K1,
                            const bf16_t* __restrict__ A2, int K2,
                            const bf16_t* __restrict__ B,
                            const float*  __restrict__ bias,
                            float*        __restrict__ C,
                            bf16_t*       __restrict__ Cbf,
                            int ldc, int ldcbf, int relu)
{
    const int lane   = threadIdx.x & 31;
    const int wave   = threadIdx.x >> 5;          // 0..1
    const int r      = lane & 15;
    const int hh     = lane >> 4;                 // K-half select (0/1)
    const int tileM  = blockIdx.x;
    const int tileO0 = blockIdx.y * 4 + wave * 2; // two adjacent O tiles
    const int K      = K1 + K2;

    const bf16_t* a1row = A1 + (size_t)(tileM * 16 + r) * K1;
    const bf16_t* b0row = B + (size_t)(tileO0 * 16 + r) * K;
    const bf16_t* b1row = b0row + (size_t)16 * K;

    v8f c0 = {}, c1 = {};

    // phase 1: K over A1
    for (int k0 = 0; k0 < K1; k0 += 32) {
        v16bf a  = load_afrag(a1row + k0, hh);
        v16bf b0 = *(const v16bf*)(b0row + k0 + hh * 16);
        v16bf b1 = *(const v16bf*)(b1row + k0 + hh * 16);
        c0 = __builtin_amdgcn_wmma_f32_16x16x32_bf16(false, a, false, b0,
                                                     (short)0, c0, false, false);
        c1 = __builtin_amdgcn_wmma_f32_16x16x32_bf16(false, a, false, b1,
                                                     (short)0, c1, false, false);
    }
    // phase 2: K over A2 (B continues at column offset K1)
    for (int k0 = 0; k0 < K2; k0 += 32) {
        const bf16_t* a2row = A2 + (size_t)(tileM * 16 + r) * K2;
        v16bf a  = load_afrag(a2row + k0, hh);
        v16bf b0 = *(const v16bf*)(b0row + K1 + k0 + hh * 16);
        v16bf b1 = *(const v16bf*)(b1row + K1 + k0 + hh * 16);
        c0 = __builtin_amdgcn_wmma_f32_16x16x32_bf16(false, a, false, b0,
                                                     (short)0, c0, false, false);
        c1 = __builtin_amdgcn_wmma_f32_16x16x32_bf16(false, a, false, b1,
                                                     (short)0, c1, false, false);
    }

#pragma unroll
    for (int j = 0; j < 2; ++j) {
        const v8f&  c    = j ? c1 : c0;
        const int   ncol = (tileO0 + j) * 16 + r;
        const float bv   = bias ? bias[ncol] : 0.0f;
#pragma unroll
        for (int i = 0; i < 8; ++i) {
            const int    mrow = tileM * 16 + i + hh * 8;
            float v = c[i] + bv;
            if (relu) v = fmaxf(v, 0.0f);
            C[(size_t)mrow * ldc + ncol] = v;
            if (Cbf) Cbf[(size_t)mrow * ldcbf + ncol] = (bf16_t)v;
        }
    }
}

// ------------------------------- helpers -----------------------------------
__device__ __forceinline__ unsigned fmap(float f) {
    int b = __float_as_int(f);
    return (unsigned)(b ^ ((b >> 31) | 0x80000000));
}
__device__ __forceinline__ float funmap(unsigned u) {
    int b = (u & 0x80000000u) ? (int)(u ^ 0x80000000u) : (int)~u;
    return __int_as_float(b);
}
__device__ __forceinline__ float sigm(float x) { return 1.0f / (1.0f + __expf(-x)); }

__global__ void k_f2bf(const float* __restrict__ s, bf16_t* __restrict__ d, long n) {
    long i = (long)blockIdx.x * blockDim.x + threadIdx.x;
    if (i < n) d[i] = (bf16_t)s[i];
}

// concat LSTM weights along K: wcat[dir*512+o][0:128]=Wih, [128:256]=Whh (bf16)
__global__ void k_wcat(const float* __restrict__ Wih, const float* __restrict__ Whh,
                       bf16_t* __restrict__ wcat, int n) {
    int i = blockIdx.x * blockDim.x + threadIdx.x;
    if (i >= n) return;
    int k = i & 255, row = i >> 8;   // row in [0, 1024)
    float v = (k < 128) ? Wih[(size_t)row * 128 + k] : Whh[(size_t)row * 128 + (k - 128)];
    wcat[i] = (bf16_t)v;
}

__global__ void k_badd(const float* __restrict__ a, const float* __restrict__ b,
                       float* __restrict__ o, int n) {
    int i = blockIdx.x * blockDim.x + threadIdx.x;
    if (i < n) o[i] = a[i] + b[i];
}

// es/ed per (node, head): 64-wide dot against attention vectors
__global__ void k_attn(const float* __restrict__ xp, const float* __restrict__ asrc,
                       const float* __restrict__ adst, float* __restrict__ es,
                       float* __restrict__ ed, int n2) {
    int i = blockIdx.x * blockDim.x + threadIdx.x;
    if (i >= n2) return;
    int node = i >> 1, h = i & 1;
    const float* row = xp + (size_t)node * 128 + h * 64;
    const float* as  = asrc + h * 64;
    const float* ad  = adst + h * 64;
    float s1 = 0.f, s2 = 0.f;
#pragma unroll 8
    for (int j = 0; j < 64; ++j) { float v = row[j]; s1 += v * as[j]; s2 += v * ad[j]; }
    es[i] = s1; ed[i] = s2;
}

__global__ void k_init_ms(unsigned* __restrict__ m, float* __restrict__ s, int n) {
    int i = blockIdx.x * blockDim.x + threadIdx.x;
    if (i < n) { m[i] = 0x007FFFFFu; /* fmap(-inf) */ s[i] = 0.0f; }
}

// pass 1: edge logits (leaky-relu) + segment max via monotone-uint atomicMax
__global__ void k_edge_max(const int* __restrict__ ei, int E, int Etot,
                           const float* __restrict__ es, const float* __restrict__ ed,
                           float* __restrict__ ebuf, unsigned* __restrict__ m) {
    int e = blockIdx.x * blockDim.x + threadIdx.x;
    if (e >= Etot) return;
    int s, d;
    if (e < E) { s = ei[e]; d = ei[E + e]; } else { s = d = e - E; }
#pragma unroll
    for (int h = 0; h < 2; ++h) {
        float v = es[s * 2 + h] + ed[d * 2 + h];
        v = v > 0.f ? v : 0.2f * v;
        ebuf[(size_t)e * 2 + h] = v;
        atomicMax(&m[d * 2 + h], fmap(v));
    }
}

// pass 2: exp(e - max) + segment sum
__global__ void k_edge_exp(const int* __restrict__ ei, int E, int Etot,
                           float* __restrict__ ebuf, const unsigned* __restrict__ m,
                           float* __restrict__ s) {
    int e = blockIdx.x * blockDim.x + threadIdx.x;
    if (e >= Etot) return;
    int d = (e < E) ? ei[E + e] : e - E;
#pragma unroll
    for (int h = 0; h < 2; ++h) {
        float ex = __expf(ebuf[(size_t)e * 2 + h] - funmap(m[d * 2 + h]));
        ebuf[(size_t)e * 2 + h] = ex;
        atomicAdd(&s[d * 2 + h], ex);
    }
}

// pass 3: out[dst] += alpha * xp[src] ; thread = (edge, 4-channel chunk)
__global__ void k_edge_aggr(const int* __restrict__ ei, int E, int Etot,
                            const float* __restrict__ ex, const float* __restrict__ s,
                            const float* __restrict__ xp, float* __restrict__ out) {
    long tid = (long)blockIdx.x * blockDim.x + threadIdx.x;
    if (tid >= (long)Etot * 32) return;
    int e = (int)(tid >> 5), q = (int)(tid & 31);
    int src, dst;
    if (e < E) { src = ei[e]; dst = ei[E + e]; } else { src = dst = e - E; }
    int h = q >> 4;
    float a = ex[(size_t)e * 2 + h] / s[dst * 2 + h];
    float4 v = *(const float4*)(xp + (size_t)src * 128 + q * 4);
    float* o = out + (size_t)dst * 128 + q * 4;
    atomicAdd(o + 0, a * v.x); atomicAdd(o + 1, a * v.y);
    atomicAdd(o + 2, a * v.z); atomicAdd(o + 3, a * v.w);
}

__global__ void k_bias_relu(float* __restrict__ X, const float* __restrict__ b,
                            bf16_t* __restrict__ Xb, int relu, int n) {
    int i = blockIdx.x * blockDim.x + threadIdx.x;
    if (i >= n) return;
    float v = X[i] + b[i & 127];
    if (relu) v = fmaxf(v, 0.f);
    X[i] = v; Xb[i] = (bf16_t)v;
}

// LSTM pointwise: gates (i,f,g,o) -> c,h ; emit fp32 h into hs slot + bf16 h
__global__ void k_lstm_gate(const float* __restrict__ g, float* __restrict__ cst,
                            float* __restrict__ hout, bf16_t* __restrict__ hbf, int n) {
    int i = blockIdx.x * blockDim.x + threadIdx.x;
    if (i >= n) return;
    int node = i >> 7, j = i & 127;
    const float* gr = g + (size_t)node * 512;
    float gi = sigm(gr[j]);
    float gf = sigm(gr[128 + j]);
    float gg = tanhf(gr[256 + j]);
    float go = sigm(gr[384 + j]);
    float c  = gf * cst[i] + gi * gg;
    cst[i]   = c;
    float h  = go * tanhf(c);
    hout[i]  = h;
    hbf[i]   = (bf16_t)h;
}

// JK attention score: dot(hf, w[:128]) + dot(hb, w[128:]) + b
__global__ void k_score(const float* __restrict__ hs, const float* __restrict__ w,
                        const float* __restrict__ bsc, float* __restrict__ score, int Nn) {
    int i = blockIdx.x * blockDim.x + threadIdx.x;
    if (i >= 3 * Nn) return;
    int t = i / Nn, n = i - t * Nn;
    const float* hf = hs + ((size_t)t * Nn + n) * 128;
    const float* hb = hs + ((size_t)(3 + t) * Nn + n) * 128;
    float acc = bsc[0];
#pragma unroll 8
    for (int k = 0; k < 128; ++k) acc += hf[k] * w[k] + hb[k] * w[128 + k];
    score[(size_t)t * Nn + n] = acc;
}

__global__ void k_softmax3(const float* __restrict__ score, float* __restrict__ al, int Nn) {
    int n = blockIdx.x * blockDim.x + threadIdx.x;
    if (n >= Nn) return;
    float s0 = score[n], s1 = score[Nn + n], s2 = score[2 * Nn + n];
    float m  = fmaxf(s0, fmaxf(s1, s2));
    float e0 = __expf(s0 - m), e1 = __expf(s1 - m), e2 = __expf(s2 - m);
    float inv = 1.0f / (e0 + e1 + e2);
    al[n] = e0 * inv; al[Nn + n] = e1 * inv; al[2 * Nn + n] = e2 * inv;
}

__global__ void k_jk(const float* __restrict__ X, const float* __restrict__ al,
                     float* __restrict__ jk, bf16_t* __restrict__ jkb, int Nn, int n) {
    int i = blockIdx.x * blockDim.x + threadIdx.x;
    if (i >= n) return;
    int node = i >> 7;
    size_t stride = (size_t)Nn * 128;
    float v = al[node] * X[i] + al[Nn + node] * X[stride + i]
            + al[2 * Nn + node] * X[2 * stride + i];
    jk[i] = v; jkb[i] = (bf16_t)v;
}

// final 6-way classifier + log_softmax (tiny: VALU)
__global__ void k_out(const float* __restrict__ h2, const float* __restrict__ ow,
                      const float* __restrict__ ob, float* __restrict__ out, int Nn) {
    int n = blockIdx.x * blockDim.x + threadIdx.x;
    if (n >= Nn) return;
    const float* h = h2 + (size_t)n * 64;
    float lg[6];
    float mx = -1e30f;
#pragma unroll
    for (int j = 0; j < 6; ++j) {
        float a = ob[j];
        for (int k = 0; k < 64; ++k) a += h[k] * ow[j * 64 + k];
        lg[j] = a; mx = fmaxf(mx, a);
    }
    float se = 0.f;
#pragma unroll
    for (int j = 0; j < 6; ++j) se += __expf(lg[j] - mx);
    float lse = mx + logf(se);
#pragma unroll
    for (int j = 0; j < 6; ++j) out[(size_t)n * 6 + j] = lg[j] - lse;
}

// ---------------------------------------------------------------------------
extern "C" void kernel_launch(void* const* d_in, const int* in_sizes, int n_in,
                              void* d_out, int out_size, void* d_ws, size_t ws_size,
                              hipStream_t stream)
{
    (void)n_in; (void)out_size; (void)ws_size;
    const float* x    = (const float*)d_in[0];
    const int*   ei   = (const int*)  d_in[1];
    const float* W1   = (const float*)d_in[2];
    const float* a1s  = (const float*)d_in[3];
    const float* a1d  = (const float*)d_in[4];
    const float* b1   = (const float*)d_in[5];
    const float* Ws   = (const float*)d_in[6];
    const float* ass  = (const float*)d_in[7];
    const float* asd  = (const float*)d_in[8];
    const float* bsv  = (const float*)d_in[9];
    const float* Wih  = (const float*)d_in[10];
    const float* Whh  = (const float*)d_in[11];
    const float* bih  = (const float*)d_in[12];
    const float* bhh  = (const float*)d_in[13];
    const float* jkw  = (const float*)d_in[14];
    const float* jkb0 = (const float*)d_in[15];
    const float* l1w  = (const float*)d_in[16];
    const float* l1b  = (const float*)d_in[17];
    const float* l2w  = (const float*)d_in[18];
    const float* l2b  = (const float*)d_in[19];
    const float* ow   = (const float*)d_in[20];
    const float* ob   = (const float*)d_in[21];

    const int Nn   = in_sizes[0] / 128;   // 50000 (divisible by 16)
    const int E    = in_sizes[1] / 2;     // 800000
    const int Etot = E + Nn;

    // workspace bump allocator
    char* wp = (char*)d_ws;
    auto alloc = [&](size_t bytes) -> void* {
        void* r = (void*)wp; wp += (bytes + 255) & ~(size_t)255; return r;
    };
    float*    Xf    = (float*)   alloc((size_t)3 * Nn * 128 * 4);
    bf16_t*   Xb    = (bf16_t*)  alloc((size_t)3 * Nn * 128 * 2);
    bf16_t*   xbf   = (bf16_t*)  alloc((size_t)Nn * 128 * 2);
    float*    xp    = (float*)   alloc((size_t)Nn * 128 * 4);
    float*    esb   = (float*)   alloc((size_t)Nn * 2 * 4);
    float*    edb   = (float*)   alloc((size_t)Nn * 2 * 4);
    unsigned* mb    = (unsigned*)alloc((size_t)Nn * 2 * 4);
    float*    sb    = (float*)   alloc((size_t)Nn * 2 * 4);
    float*    ebuf  = (float*)   alloc((size_t)Etot * 2 * 4);
    bf16_t*   w1bf  = (bf16_t*)  alloc((size_t)128 * 128 * 2);
    bf16_t*   wsbf  = (bf16_t*)  alloc((size_t)2 * 128 * 128 * 2);
    bf16_t*   wcat  = (bf16_t*)  alloc((size_t)2 * 512 * 256 * 2);  // [dir][512][256]
    float*    bsum  = (float*)   alloc((size_t)2 * 512 * 4);
    bf16_t*   l1bf  = (bf16_t*)  alloc((size_t)64 * 128 * 2);
    bf16_t*   l2bf  = (bf16_t*)  alloc((size_t)64 * 192 * 2);
    float*    gbuf  = (float*)   alloc((size_t)Nn * 512 * 4);
    float*    cst   = (float*)   alloc((size_t)Nn * 128 * 4);
    bf16_t*   hbf   = (bf16_t*)  alloc((size_t)Nn * 128 * 2);
    float*    hs    = (float*)   alloc((size_t)6 * Nn * 128 * 4);   // [dir][t][N][128]
    float*    score = (float*)   alloc((size_t)3 * Nn * 4);
    float*    alp   = (float*)   alloc((size_t)3 * Nn * 4);
    float*    jkf   = (float*)   alloc((size_t)Nn * 128 * 4);
    bf16_t*   jkbfb = (bf16_t*)  alloc((size_t)Nn * 128 * 2);
    bf16_t*   h1bf  = (bf16_t*)  alloc((size_t)Nn * 64 * 2);
    float*    h2    = (float*)   alloc((size_t)Nn * 64 * 4);

    auto conv = [&](const float* s, bf16_t* d, long n) {
        k_f2bf<<<(unsigned)((n + 255) / 256), 256, 0, stream>>>(s, d, n);
    };
    // C[M,O] = [A1|A2] @ B^T, O cols per block = 64, 2 waves x 2 O-tiles
    auto gemm = [&](const bf16_t* A1, int K1, const bf16_t* A2, int K2,
                    const bf16_t* B, const float* bias, float* C, bf16_t* Cbf,
                    int O, int ldc, int ldcbf, int relu) {
        dim3 g(Nn / 16, O / 64);
        k_gemm_bf16<<<g, 64, 0, stream>>>(A1, K1, A2, K2, B, bias, C, Cbf,
                                          ldc, ldcbf, relu);
    };

    // one-time (per call) weight prep
    conv(x,  xbf,  (long)Nn * 128);
    conv(W1, w1bf, 128 * 128);
    conv(Ws, wsbf, 2 * 128 * 128);
    conv(l1w, l1bf, 64 * 128);
    conv(l2w, l2bf, 64 * 192);
    k_wcat<<<(2 * 512 * 256 + 255) / 256, 256, 0, stream>>>(Wih, Whh, wcat, 2 * 512 * 256);
    k_badd<<<(2 * 512 + 255) / 256, 256, 0, stream>>>(bih, bhh, bsum, 2 * 512);

    // ---------------- 3 GAT layers ----------------
    const bf16_t* inbf = xbf;
    for (int l = 0; l < 3; ++l) {
        const bf16_t* Wb = (l == 0) ? w1bf : wsbf + (size_t)(l - 1) * 128 * 128;
        const float*  as = (l == 0) ? a1s  : ass + (size_t)(l - 1) * 128;
        const float*  ad = (l == 0) ? a1d  : asd + (size_t)(l - 1) * 128;
        const float*  bb = (l == 0) ? b1   : bsv + (size_t)(l - 1) * 128;

        gemm(inbf, 128, nullptr, 0, Wb, nullptr, xp, nullptr, 128, 128, 0, 0);
        k_attn<<<(2 * Nn + 255) / 256, 256, 0, stream>>>(xp, as, ad, esb, edb, 2 * Nn);
        k_init_ms<<<(2 * Nn + 255) / 256, 256, 0, stream>>>(mb, sb, 2 * Nn);
        float* Xl = Xf + (size_t)l * Nn * 128;
        hipMemsetAsync(Xl, 0, (size_t)Nn * 128 * 4, stream);
        k_edge_max<<<(Etot + 255) / 256, 256, 0, stream>>>(ei, E, Etot, esb, edb, ebuf, mb);
        k_edge_exp<<<(Etot + 255) / 256, 256, 0, stream>>>(ei, E, Etot, ebuf, mb, sb);
        long aggT = (long)Etot * 32;
        k_edge_aggr<<<(unsigned)((aggT + 255) / 256), 256, 0, stream>>>(ei, E, Etot, ebuf, sb, xp, Xl);
        bf16_t* Xlb = Xb + (size_t)l * Nn * 128;
        k_bias_relu<<<(Nn * 128 + 255) / 256, 256, 0, stream>>>(Xl, bb, Xlb, l > 0, Nn * 128);
        inbf = Xlb;
    }

    // ---------------- JK bi-LSTM over layer axis ----------------
    // single fused GEMM per step: g = [X[t] | h] @ [Wih | Whh]^T + (bih+bhh)
    for (int d = 0; d < 2; ++d) {
        hipMemsetAsync(cst, 0, (size_t)Nn * 128 * 4, stream);
        hipMemsetAsync(hbf, 0, (size_t)Nn * 128 * 2, stream);
        for (int step = 0; step < 3; ++step) {
            int t = (d == 0) ? step : 2 - step;
            gemm(Xb + (size_t)t * Nn * 128, 128, hbf, 128,
                 wcat + (size_t)d * 512 * 256, bsum + (size_t)d * 512,
                 gbuf, nullptr, 512, 512, 0, 0);
            k_lstm_gate<<<(Nn * 128 + 255) / 256, 256, 0, stream>>>(
                gbuf, cst, hs + ((size_t)d * 3 + t) * Nn * 128, hbf, Nn * 128);
        }
    }

    // attention over layers + weighted sum
    k_score<<<(3 * Nn + 255) / 256, 256, 0, stream>>>(hs, jkw, jkb0, score, Nn);
    k_softmax3<<<(Nn + 255) / 256, 256, 0, stream>>>(score, alp, Nn);
    k_jk<<<(Nn * 128 + 255) / 256, 256, 0, stream>>>(Xf, alp, jkf, jkbfb, Nn, Nn * 128);

    // ---------------- MLP head ----------------
    // h1 = relu(jk @ lin1^T + b), bf16 mirror -> h1bf
    gemm(jkbfb, 128, nullptr, 0, l1bf, l1b, xp /*scratch*/, h1bf, 64, 64, 64, 1);
    // h2 = relu([h1 | x] @ lin2^T + b)  (split-K: no concat buffer needed)
    gemm(h1bf, 64, xbf, 128, l2bf, l2b, h2, nullptr, 64, 64, 0, 1);
    // logits + log_softmax
    k_out<<<(Nn + 255) / 256, 256, 0, stream>>>(h2, ow, ob, (float*)d_out, Nn);
}
